// Model_28793460753003
// MI455X (gfx1250) — matrix operand order repaired and checked by reading, not compiled, not run on hardware
//
#include <hip/hip_runtime.h>
#include <math.h>
#include <float.h>
#include <cmath>

// ---------------- model constants ----------------
constexpr int BATCH = 8;
constexpr int SEQ   = 1024;   // encoder length
constexpr int DECL  = 512;    // decoder length (LABEL+PRED)
constexpr int DM    = 512;    // d_model
constexpr int NH    = 8;      // heads
constexpr int HD    = 64;     // head dim
constexpr int DFF   = 2048;
constexpr int COUT  = 7;
constexpr int PRED  = 256;
constexpr int LABEL = 256;

typedef __attribute__((ext_vector_type(16))) _Float16 v16h;
typedef __attribute__((ext_vector_type(8)))  float    v8f;
typedef __attribute__((ext_vector_type(4)))  unsigned v4u;
typedef __attribute__((ext_vector_type(8)))  int      v8i_t;

union AFrag { v16h v; _Float16 h[16]; };
union CFrag { v8f  v; float    f[8];  };

// ---------------- deterministic sampling hash ----------------
__device__ __forceinline__ unsigned hash3(unsigned seed, unsigned l, unsigned s) {
    unsigned x = seed * 0x9E3779B9u ^ (l * 0x85EBCA6Bu) ^ (s * 0xC2B2AE35u);
    x ^= x >> 16; x *= 0x7FEB352Du;
    x ^= x >> 15; x *= 0x846CA68Bu;
    x ^= x >> 16;
    return x;
}

// ---------------- Tensor Data Mover: 2-D tile load into LDS ----------------
// Builds a D# (groups 0+1, VADDR2/3 = NULL -> tensors up to 2D) per
// CDNA5 ISA and issues tensor_load_to_lds.
// g1w0 carries data_size / pad controls (group1 dword0).
__device__ __forceinline__ void tdm_load_2d(const void* gaddr, unsigned ldsOff,
                                            unsigned g1w0,
                                            unsigned tile_d0, unsigned tile_d1,
                                            unsigned tdim0, unsigned tdim1,
                                            unsigned stride0) {
    unsigned long long ga = (unsigned long long)gaddr;   // byte address of tile start
    v4u g0;
    g0.x = 1u;                                           // count=1, user mode
    g0.y = ldsOff;                                       // lds_addr (bytes)
    g0.z = (unsigned)ga;                                 // global_addr[31:0]
    g0.w = (unsigned)((ga >> 32) & 0x01FFFFFFull)        // global_addr[56:32]
         | (2u << 30);                                   // type = 2 ("image")
    v8i_t g1;
    g1[0] = (int)g1w0;                                   // mask=0 | data_size | pad ctl
    g1[1] = (int)((tdim0 & 0xFFFFu) << 16);              // tensor_dim0[15:0] @ bits63:48
    g1[2] = (int)((tdim0 >> 16) | ((tdim1 & 0xFFFFu) << 16));
    g1[3] = (int)((tdim1 >> 16) | (tile_d0 << 16));      // tile_dim0 @ bits127:112
    g1[4] = (int)tile_d1;                                // tile_dim1 (tile_dim2 = 0)
    g1[5] = (int)stride0;                                // tensor_dim0_stride[31:0]
    g1[6] = 0;                                           // stride0 hi | dim1_stride lo
    g1[7] = 0;
    asm volatile("tensor_load_to_lds %0, %1" :: "s"(g0), "s"(g1) : "memory");
}

// low 32 bits of a flat LDS address are the LDS byte offset (ISA §10.2 aperture map)
__device__ __forceinline__ unsigned lds_off(const void* p) {
    return (unsigned)(unsigned long long)p;
}

// ---------------- per-(b,c) mean/std over SEQ ----------------
__global__ void __launch_bounds__(256)
stats_kernel(const float* __restrict__ x, float* __restrict__ mean,
             float* __restrict__ stdv, int L, int C) {
    int bc = blockIdx.x;          // b*C + c
    int b = bc / C, c = bc % C;
    __shared__ float s1[256], s2[256];
    float sum = 0.f, sq = 0.f;
    for (int l = threadIdx.x; l < L; l += 256) {
        float v = x[(size_t)(b * L + l) * C + c];
        sum += v; sq += v * v;
    }
    s1[threadIdx.x] = sum; s2[threadIdx.x] = sq; __syncthreads();
    for (int o = 128; o > 0; o >>= 1) {
        if (threadIdx.x < o) { s1[threadIdx.x] += s1[threadIdx.x + o]; s2[threadIdx.x] += s2[threadIdx.x + o]; }
        __syncthreads();
    }
    if (threadIdx.x == 0) {
        float m = s1[0] / (float)L;
        float var = s2[0] / (float)L - m * m;
        mean[bc] = m;
        stdv[bc] = sqrtf(var + 1e-5f);
    }
}

__global__ void __launch_bounds__(256)
normalize_kernel(const float* __restrict__ x, const float* __restrict__ mean,
                 const float* __restrict__ stdv, float* __restrict__ xc) {
    int t = blockIdx.x * 256 + threadIdx.x;
    if (t >= BATCH * SEQ * COUT) return;
    int c = t % COUT;
    int b = t / (SEQ * COUT);
    xc[t] = (x[t] - mean[b * COUT + c]) / stdv[b * COUT + c];
}

// ---------------- data embedding: circular conv + mark + positional ----------------
__global__ void __launch_bounds__(256)
embed_kernel(const float* __restrict__ x, int Cin,
             const float* __restrict__ xmark,
             const float* __restrict__ convw,   // (DM, Cin, 3)
             const float* __restrict__ tempw,   // (4, DM)
             float* __restrict__ out, int L) {
    int t = blockIdx.x * 256 + threadIdx.x;
    if (t >= BATCH * L * DM) return;
    int d = t % DM;
    int l = (t / DM) % L;
    int b = t / (DM * L);
    float acc = 0.f;
    #pragma unroll
    for (int k = 0; k < 3; ++k) {
        int ls = l + k - 1;
        ls = (ls + L) % L;                      // circular pad
        const float* xp = x + (size_t)(b * L + ls) * Cin;
        const float* w  = convw + (size_t)d * Cin * 3 + k;
        for (int c = 0; c < Cin; ++c) acc += xp[c] * w[c * 3];
    }
    const float* mk = xmark + (size_t)(b * L + l) * 4;
    #pragma unroll
    for (int m = 0; m < 4; ++m) acc += mk[m] * tempw[m * DM + d];
    // positional encoding
    int i2 = (d >> 1) * 2;
    float freq = expf(-(float)i2 * (logf(10000.f) / (float)DM));
    float ang  = (float)l * freq;
    acc += (d & 1) ? cosf(ang) : sinf(ang);
    out[t] = acc;
}

// ---------------- TDM + WMMA GEMM: C[M,N] = A[M,K] @ W[K,N] + bias (opt GELU) ----
// Block tile 128x64, 8 waves; wave w computes rows [16w,16w+16) x 64 cols
// (4 accumulators, A fragment reused across 4 v_wmma_f32_16x16x32_f16).
// A/B fp32 tiles staged into LDS by the Tensor Data Mover (double buffered);
// A rows padded by TDM (2 dwords / 32) -> LDS stride 34 for bank spread and
// 8-byte-aligned ds_load_b64 fragment reads.
// Fragment layouts per CDNA5 ISA §7.12.2 (wave32).
constexpr int GBM = 128, GBN = 64, GBK = 32;
constexpr int LDA = 34;                        // padded A row stride (floats)
constexpr int SA_BUF = GBM * LDA;              // 4352 floats / buffer
constexpr int SB_BUF = GBK * GBN;              // 2048 floats / buffer

__global__ void __launch_bounds__(256)
gemm_wmma_kernel(const float* __restrict__ A, const float* __restrict__ W,
                 const float* __restrict__ bias, float* __restrict__ C,
                 int M, int N, int K, int act) {
    __shared__ float smem[2 * SA_BUF + 2 * SB_BUF];

    const int wave  = threadIdx.x >> 5;
    const int lane  = threadIdx.x & 31;
    const int l15   = lane & 15;
    const int khalf = lane >> 4;

    const int ntile = N / GBN;
    const int bm = blockIdx.x / ntile;
    const int bn = blockIdx.x % ntile;
    const int row0 = bm * GBM;
    const int col0 = bn * GBN;
    const int nsteps = K / GBK;

    const unsigned base  = lds_off(smem);
    // group1 dword0: data_size=4B (code 2 @ [17:16]); A adds pad_enable(20),
    // pad_interval=4 (32 dwords, [24:22]), pad_amount=1 (2 dwords, [31:25]).
    const unsigned G1A = (2u << 16) | (1u << 20) | (4u << 22) | (1u << 25);
    const unsigned G1B = (2u << 16);

    CFrag acc[4];
    #pragma unroll
    for (int nt = 0; nt < 4; ++nt)
        #pragma unroll
        for (int r = 0; r < 8; ++r) acc[nt].f[r] = 0.f;

    // prologue: DMA tiles for k-step 0 into buffer 0
    if (wave == 0) {
        tdm_load_2d(A + (size_t)row0 * K, base, G1A,
                    GBK, GBM, (unsigned)K, (unsigned)M, (unsigned)K);
        tdm_load_2d(W + col0, base + 2u * SA_BUF * 4u, G1B,
                    GBN, GBK, (unsigned)N, (unsigned)K, (unsigned)N);
    }

    int buf = 0;
    for (int t = 0; t < nsteps; ++t) {
        if (wave == 0) __builtin_amdgcn_s_wait_tensorcnt(0);  // DMA for tile t done
        __syncthreads();   // publish tile t; all waves finished reading tile t-1
        if (wave == 0 && t + 1 < nsteps) {                    // prefetch tile t+1
            int nb = buf ^ 1;
            tdm_load_2d(A + (size_t)row0 * K + (size_t)(t + 1) * GBK,
                        base + (unsigned)nb * SA_BUF * 4u, G1A,
                        GBK, GBM, (unsigned)K, (unsigned)M, (unsigned)K);
            tdm_load_2d(W + (size_t)(t + 1) * GBK * N + col0,
                        base + 2u * SA_BUF * 4u + (unsigned)nb * SB_BUF * 4u, G1B,
                        GBN, GBK, (unsigned)N, (unsigned)K, (unsigned)N);
        }

        const float* sa  = smem + buf * SA_BUF + (wave * 16 + l15) * LDA;
        const float* sbb = smem + 2 * SA_BUF + buf * SB_BUF;

        AFrag a;
        #pragma unroll
        for (int vv = 0; vv < 8; ++vv) {
            int ka = ((vv < 4) ? 0 : 16) + 8 * khalf + 2 * (vv & 3);
            float2 av = *reinterpret_cast<const float2*>(sa + ka);   // ds_load_b64
            a.h[2 * vv]     = (_Float16)av.x;
            a.h[2 * vv + 1] = (_Float16)av.y;
        }
        #pragma unroll
        for (int nt = 0; nt < 4; ++nt) {
            AFrag b;
            const float* sc = sbb + nt * 16 + l15;
            #pragma unroll
            for (int vv = 0; vv < 8; ++vv) {
                int kb = ((vv < 4) ? 0 : 16) + 8 * khalf + 2 * (vv & 3);
                b.h[2 * vv]     = (_Float16)sc[kb * GBN];
                b.h[2 * vv + 1] = (_Float16)sc[(kb + 1) * GBN];
            }
            acc[nt].v = __builtin_amdgcn_wmma_f32_16x16x32_f16(
                false, a.v, false, b.v, (short)0, acc[nt].v, false, false);
        }
        buf ^= 1;
    }

    #pragma unroll
    for (int nt = 0; nt < 4; ++nt) {
        int col = col0 + nt * 16 + l15;
        float bv = bias ? bias[col] : 0.f;
        #pragma unroll
        for (int r = 0; r < 8; ++r) {
            int row = row0 + wave * 16 + r + 8 * khalf;
            float v = acc[nt].f[r] + bv;
            if (act == 1) v = 0.5f * v * (1.f + erff(v * 0.70710678f));  // exact GELU
            C[(size_t)row * N + col] = v;
        }
    }
}

// ---------------- V mean over keys (non-masked context) ----------------
__global__ void __launch_bounds__(256)
vmean_kernel(const float* __restrict__ V, float* __restrict__ vmean, int Lk) {
    int t = blockIdx.x * 256 + threadIdx.x;
    if (t >= BATCH * NH * HD) return;
    int e = t % HD;
    int h = (t / HD) % NH;
    int b = t / (HD * NH);
    float s = 0.f;
    for (int l = 0; l < Lk; ++l) s += V[(size_t)(b * Lk + l) * DM + h * HD + e];
    vmean[t] = s / (float)Lk;
}

__global__ void __launch_bounds__(256)
ctx_mean_kernel(float* __restrict__ ctx, const float* __restrict__ vmean, int Lq) {
    int t = blockIdx.x * 256 + threadIdx.x;
    if (t >= BATCH * NH * Lq * HD) return;
    int e = t % HD;
    int h = (t / (HD * Lq)) % NH;
    int b = t / (HD * Lq * NH);
    ctx[t] = vmean[(b * NH + h) * HD + e];
}

// ---------------- cumsum(V) context (masked / decoder self-attn) ----------------
__global__ void __launch_bounds__(256)
ctx_cumsum_kernel(float* __restrict__ ctx, const float* __restrict__ V, int Lq) {
    int t = blockIdx.x * 256 + threadIdx.x;
    if (t >= BATCH * NH * HD) return;
    int e = t % HD;
    int h = (t / HD) % NH;
    int b = t / (HD * NH);
    float acc = 0.f;
    for (int l = 0; l < Lq; ++l) {
        acc += V[(size_t)(b * Lq + l) * DM + h * HD + e];
        ctx[((size_t)(b * NH + h) * Lq + l) * HD + e] = acc;
    }
}

// ---------------- sparsity metric: M = max_s(QK_s) - sum_s(QK_s)/Lk ----------------
__global__ void __launch_bounds__(256)
metric_kernel(const float* __restrict__ Q, const float* __restrict__ K,
              float* __restrict__ metric, int Lq, int Lk, int U, int seed) {
    int t = blockIdx.x * 256 + threadIdx.x;
    if (t >= BATCH * NH * Lq) return;
    int l = t % Lq;
    int h = (t / Lq) % NH;
    int b = t / (Lq * NH);
    const float* q = Q + (size_t)(b * Lq + l) * DM + h * HD;
    float mx = -FLT_MAX, sm = 0.f;
    for (int s = 0; s < U; ++s) {
        int j = (int)(hash3((unsigned)seed, (unsigned)l, (unsigned)s) % (unsigned)Lk);
        const float* kk = K + (size_t)(b * Lk + j) * DM + h * HD;
        float d = 0.f;
        for (int e = 0; e < HD; ++e) d += q[e] * kk[e];
        mx = fmaxf(mx, d);
        sm += d;
    }
    metric[(b * NH + h) * Lq + l] = mx - sm / (float)Lk;
}

// ---------------- top-u selection per (b,h): iterative argmax ----------------
__global__ void __launch_bounds__(256)
topk_kernel(const float* __restrict__ metric, int* __restrict__ topk, int Lq, int u) {
    __shared__ float vals[1024];
    __shared__ float sval[256];
    __shared__ int   sidx[256];
    int bh = blockIdx.x;
    for (int l = threadIdx.x; l < Lq; l += 256) vals[l] = metric[bh * Lq + l];
    __syncthreads();
    for (int it = 0; it < u; ++it) {
        float best = -FLT_MAX; int bi = 0x7fffffff;
        for (int l = threadIdx.x; l < Lq; l += 256) {
            float v = vals[l];
            if (v > best || (v == best && l < bi)) { best = v; bi = l; }
        }
        sval[threadIdx.x] = best; sidx[threadIdx.x] = bi; __syncthreads();
        for (int o = 128; o > 0; o >>= 1) {
            if (threadIdx.x < o) {
                float v2 = sval[threadIdx.x + o]; int i2 = sidx[threadIdx.x + o];
                if (v2 > sval[threadIdx.x] || (v2 == sval[threadIdx.x] && i2 < sidx[threadIdx.x])) {
                    sval[threadIdx.x] = v2; sidx[threadIdx.x] = i2;
                }
            }
            __syncthreads();
        }
        if (threadIdx.x == 0) { topk[bh * u + it] = sidx[0]; vals[sidx[0]] = -FLT_MAX; }
        __syncthreads();
    }
}

// ---------------- scores + softmax + attn@V scatter into context ----------------
__global__ void __launch_bounds__(256)
attn_update_kernel(const float* __restrict__ Q, const float* __restrict__ K,
                   const float* __restrict__ V, const int* __restrict__ topk,
                   float* __restrict__ ctx, int Lq, int Lk, int u, int mask) {
    __shared__ float q[HD];
    __shared__ float sc[1024];
    __shared__ float red[256];
    int blk = blockIdx.x;
    int ui = blk % u;
    int h  = (blk / u) % NH;
    int b  = blk / (u * NH);
    int lq = topk[(b * NH + h) * u + ui];
    if (threadIdx.x < HD) q[threadIdx.x] = Q[(size_t)(b * Lq + lq) * DM + h * HD + threadIdx.x];
    __syncthreads();

    float lmax = -FLT_MAX;
    for (int j = threadIdx.x; j < Lk; j += 256) {
        const float* kk = K + (size_t)(b * Lk + j) * DM + h * HD;
        float d = 0.f;
        for (int e = 0; e < HD; ++e) d += q[e] * kk[e];
        d *= 0.125f;                               // 1/sqrt(64)
        if (mask && j > lq) d = -FLT_MAX;          // causal
        sc[j] = d;
        lmax = fmaxf(lmax, d);
    }
    red[threadIdx.x] = lmax; __syncthreads();
    for (int o = 128; o > 0; o >>= 1) {
        if (threadIdx.x < o) red[threadIdx.x] = fmaxf(red[threadIdx.x], red[threadIdx.x + o]);
        __syncthreads();
    }
    float mx = red[0]; __syncthreads();

    float lsum = 0.f;
    for (int j = threadIdx.x; j < Lk; j += 256) {
        float p = __expf(sc[j] - mx);
        sc[j] = p;
        lsum += p;
    }
    red[threadIdx.x] = lsum; __syncthreads();
    for (int o = 128; o > 0; o >>= 1) {
        if (threadIdx.x < o) red[threadIdx.x] += red[threadIdx.x + o];
        __syncthreads();
    }
    float inv = 1.f / red[0];
    __syncthreads();

    if (threadIdx.x < HD) {
        int e = threadIdx.x;
        float acc = 0.f;
        for (int j = 0; j < Lk; ++j) acc += sc[j] * V[(size_t)(b * Lk + j) * DM + h * HD + e];
        ctx[((size_t)(b * NH + h) * Lq + lq) * HD + e] = acc * inv;
    }
}

// ---------------- (B,H,L,E) -> (B*L, DM) head merge ----------------
__global__ void __launch_bounds__(256)
merge_heads_kernel(const float* __restrict__ ctx, float* __restrict__ out, int Lq) {
    int t = blockIdx.x * 256 + threadIdx.x;
    if (t >= BATCH * Lq * DM) return;
    int d = t % DM;
    int l = (t / DM) % Lq;
    int b = t / (DM * Lq);
    out[t] = ctx[((size_t)(b * NH + d / HD) * Lq + l) * HD + (d % HD)];
}

// ---------------- out = LayerNorm(x [+ y]) * g + b, row width DM=512 ----------------
__global__ void __launch_bounds__(256)
add_ln_kernel(const float* __restrict__ x, const float* __restrict__ y,
              const float* __restrict__ g, const float* __restrict__ bt,
              float* __restrict__ out) {
    __shared__ float red[256];
    __shared__ float stat[2];
    size_t r = blockIdx.x;
    int t = threadIdx.x;
    const float* xr = x + r * DM;
    float v0 = xr[t], v1 = xr[t + 256];
    if (y) { const float* yr = y + r * DM; v0 += yr[t]; v1 += yr[t + 256]; }
    red[t] = v0 + v1; __syncthreads();
    for (int o = 128; o > 0; o >>= 1) { if (t < o) red[t] += red[t + o]; __syncthreads(); }
    if (t == 0) stat[0] = red[0] / (float)DM;
    __syncthreads();
    float m = stat[0];
    float d0 = v0 - m, d1 = v1 - m;
    red[t] = d0 * d0 + d1 * d1; __syncthreads();
    for (int o = 128; o > 0; o >>= 1) { if (t < o) red[t] += red[t + o]; __syncthreads(); }
    if (t == 0) stat[1] = 1.f / sqrtf(red[0] / (float)DM + 1e-5f);
    __syncthreads();
    float rs = stat[1];
    float* orow = out + r * DM;
    orow[t]       = d0 * rs * g[t]       + bt[t];
    orow[t + 256] = d1 * rs * g[t + 256] + bt[t + 256];
}

// ---------------- final projection DM -> COUT (tiny, VALU) ----------------
__global__ void __launch_bounds__(256)
proj_kernel(const float* __restrict__ x, const float* __restrict__ w,
            const float* __restrict__ bv, float* __restrict__ out, int rows) {
    int t = blockIdx.x * 256 + threadIdx.x;
    if (t >= rows * COUT) return;
    int c = t % COUT;
    int r = t / COUT;
    const float* xr = x + (size_t)r * DM;
    float acc = bv[c];
    for (int k = 0; k < DM; ++k) acc += xr[k] * w[k * COUT + c];
    out[t] = acc;
}

// ---------------- de-normalize + slice last PRED steps ----------------
__global__ void __launch_bounds__(256)
final_kernel(const float* __restrict__ proj, const float* __restrict__ mean,
             const float* __restrict__ stdv, float* __restrict__ out) {
    int t = blockIdx.x * 256 + threadIdx.x;
    if (t >= BATCH * PRED * COUT) return;
    int c  = t % COUT;
    int tt = (t / COUT) % PRED;
    int b  = t / (COUT * PRED);
    float v = proj[(size_t)(b * DECL + LABEL + tt) * COUT + c];
    out[t] = v * stdv[b * COUT + c] + mean[b * COUT + c];
}

// =================================================================
// Host orchestration
// =================================================================
struct AttnP { const float *wq,*bq,*wk,*bk,*wv,*bv,*wo,*bo; };

static inline int cdiv(int a, int b) { return (a + b - 1) / b; }

static void launch_gemm(const float* A, const float* W, const float* b, float* C,
                        int M, int N, int K, int act, hipStream_t s) {
    int blocks = (M / GBM) * (N / GBN);    // all shapes divide exactly
    gemm_wmma_kernel<<<blocks, 256, 0, s>>>(A, W, b, C, M, N, K, act);
}

static void run_attn(const float* xq, int Lq, const float* xkv, int Lk,
                     const AttnP& p, bool mask, int seed,
                     float* Qb, float* Kb, float* Vb, float* CTX, float* MRG, float* OUT,
                     float* vmean, float* metric, int* topk, hipStream_t s) {
    int Mq = BATCH * Lq, Mk = BATCH * Lk;
    launch_gemm(xq,  p.wq, p.bq, Qb, Mq, DM, DM, 0, s);
    launch_gemm(xkv, p.wk, p.bk, Kb, Mk, DM, DM, 0, s);
    launch_gemm(xkv, p.wv, p.bv, Vb, Mk, DM, DM, 0, s);

    int U = 5 * (int)std::ceil(std::log((double)Lk)); if (U > Lk) U = Lk;
    int u = 5 * (int)std::ceil(std::log((double)Lq)); if (u > Lq) u = Lq;

    if (mask) {
        ctx_cumsum_kernel<<<cdiv(BATCH * NH * HD, 256), 256, 0, s>>>(CTX, Vb, Lq);
    } else {
        vmean_kernel<<<cdiv(BATCH * NH * HD, 256), 256, 0, s>>>(Vb, vmean, Lk);
        ctx_mean_kernel<<<cdiv(BATCH * NH * Lq * HD, 256), 256, 0, s>>>(CTX, vmean, Lq);
    }
    metric_kernel<<<cdiv(BATCH * NH * Lq, 256), 256, 0, s>>>(Qb, Kb, metric, Lq, Lk, U, seed);
    topk_kernel<<<BATCH * NH, 256, 0, s>>>(metric, topk, Lq, u);
    attn_update_kernel<<<BATCH * NH * u, 256, 0, s>>>(Qb, Kb, Vb, topk, CTX, Lq, Lk, u, mask ? 1 : 0);
    merge_heads_kernel<<<cdiv(BATCH * Lq * DM, 256), 256, 0, s>>>(CTX, MRG, Lq);
    launch_gemm(MRG, p.wo, p.bo, OUT, Mq, DM, DM, 0, s);
}

extern "C" void kernel_launch(void* const* d_in, const int* in_sizes, int n_in,
                              void* d_out, int out_size, void* d_ws, size_t ws_size,
                              hipStream_t stream) {
    (void)in_sizes; (void)n_in; (void)out_size; (void)ws_size;

    // ---- inputs in setup_inputs() insertion order, params recursively ----
    int ii = 0;
    auto F = [&](void) -> const float* { return (const float*)d_in[ii++]; };
    const float* x_enc      = F();
    const float* x_mark_enc = F();
    const float* x_dec      = F();
    const float* x_mark_dec = F();
    const float* emb_enc_conv = F();
    const float* emb_enc_temp = F();
    const float* emb_dec_conv = F();
    const float* emb_dec_temp = F();

    auto parse_attn = [&](AttnP& a) {
        a.wq = F(); a.bq = F(); a.wk = F(); a.bk = F();
        a.wv = F(); a.bv = F(); a.wo = F(); a.bo = F();
    };
    struct EncL { AttnP attn; const float *w1,*b1,*w2,*b2,*n1g,*n1b,*n2g,*n2b; } encL[2];
    for (int i = 0; i < 2; ++i) {
        parse_attn(encL[i].attn);
        encL[i].w1 = F(); encL[i].b1 = F(); encL[i].w2 = F(); encL[i].b2 = F();
        encL[i].n1g = F(); encL[i].n1b = F(); encL[i].n2g = F(); encL[i].n2b = F();
    }
    const float* enc_ng = F();
    const float* enc_nb = F();
    struct DecL { AttnP self, cross; const float *w1,*b1,*w2,*b2,*n1g,*n1b,*n2g,*n2b,*n3g,*n3b; } decL;
    parse_attn(decL.self);
    parse_attn(decL.cross);
    decL.w1 = F(); decL.b1 = F(); decL.w2 = F(); decL.b2 = F();
    decL.n1g = F(); decL.n1b = F(); decL.n2g = F(); decL.n2b = F();
    decL.n3g = F(); decL.n3b = F();
    const float* dec_ng = F();
    const float* dec_nb = F();
    const float* proj_w = F();
    const float* proj_b = F();

    // ---- workspace layout (floats) ----
    const size_t S = (size_t)BATCH * SEQ * DM;   // 4,194,304
    float* ws   = (float*)d_ws;
    float* X    = ws + 0 * S;            // encoder activations / final enc out
    float* Q    = ws + 1 * S;
    float* Kb   = ws + 2 * S;
    float* Vb   = ws + 3 * S;
    float* CTX  = ws + 4 * S;
    float* FF   = ws + 1 * S;            // FFN hidden overlaps dead Q/K/V/CTX (4S)
    float* MRG  = ws + 5 * S;
    float* T1   = ws + 6 * S;
    float* DX   = ws + 7 * S;            // decoder activations (0.5S)
    float* MISC = ws + 7 * S + S / 2;
    float* meanb  = MISC + 0;
    float* stdb   = MISC + 64;
    float* vmean  = MISC + 128;
    float* metric = MISC + 4224;
    float* projo  = MISC + 69760;
    float* xc     = MISC + 98432;
    int*   topk   = (int*)(MISC + 155776);
    float* outp   = (float*)d_out;

    // ---- 1. series normalization stats + normalized encoder input ----
    stats_kernel<<<BATCH * COUT, 256, 0, stream>>>(x_enc, meanb, stdb, SEQ, COUT);
    normalize_kernel<<<cdiv(BATCH * SEQ * COUT, 256), 256, 0, stream>>>(x_enc, meanb, stdb, xc);

    // ---- 2. encoder embedding ----
    embed_kernel<<<cdiv(BATCH * SEQ * DM, 256), 256, 0, stream>>>(
        xc, COUT, x_mark_enc, emb_enc_conv, emb_enc_temp, X, SEQ);

    // ---- 3. encoder layers ----
    for (int i = 0; i < 2; ++i) {
        run_attn(X, SEQ, X, SEQ, encL[i].attn, false, i,
                 Q, Kb, Vb, CTX, MRG, T1, vmean, metric, topk, stream);
        add_ln_kernel<<<BATCH * SEQ, 256, 0, stream>>>(X, T1, encL[i].n1g, encL[i].n1b, X);
        launch_gemm(X,  encL[i].w1, encL[i].b1, FF, BATCH * SEQ, DFF, DM, 1, stream);   // GELU
        launch_gemm(FF, encL[i].w2, encL[i].b2, T1, BATCH * SEQ, DM, DFF, 0, stream);
        add_ln_kernel<<<BATCH * SEQ, 256, 0, stream>>>(X, T1, encL[i].n2g, encL[i].n2b, X);
    }
    add_ln_kernel<<<BATCH * SEQ, 256, 0, stream>>>(X, nullptr, enc_ng, enc_nb, X);

    // ---- 4. decoder embedding ----
    embed_kernel<<<cdiv(BATCH * DECL * DM, 256), 256, 0, stream>>>(
        x_dec, COUT, x_mark_dec, emb_dec_conv, emb_dec_temp, DX, DECL);

    // ---- 5. decoder layer ----
    run_attn(DX, DECL, DX, DECL, decL.self, true, 1000,
             Q, Kb, Vb, CTX, MRG, T1, vmean, metric, topk, stream);
    add_ln_kernel<<<BATCH * DECL, 256, 0, stream>>>(DX, T1, decL.n1g, decL.n1b, DX);

    run_attn(DX, DECL, X, SEQ, decL.cross, false, 1100,
             Q, Kb, Vb, CTX, MRG, T1, vmean, metric, topk, stream);
    add_ln_kernel<<<BATCH * DECL, 256, 0, stream>>>(DX, T1, decL.n2g, decL.n2b, DX);

    launch_gemm(DX, decL.w1, decL.b1, FF, BATCH * DECL, DFF, DM, 1, stream);
    launch_gemm(FF, decL.w2, decL.b2, T1, BATCH * DECL, DM, DFF, 0, stream);
    add_ln_kernel<<<BATCH * DECL, 256, 0, stream>>>(DX, T1, decL.n3g, decL.n3b, DX);

    add_ln_kernel<<<BATCH * DECL, 256, 0, stream>>>(DX, nullptr, dec_ng, dec_nb, DX);

    // ---- 6. output projection + de-normalization ----
    proj_kernel<<<cdiv(BATCH * DECL * COUT, 256), 256, 0, stream>>>(DX, proj_w, proj_b, projo, BATCH * DECL);
    final_kernel<<<cdiv(BATCH * PRED * COUT, 256), 256, 0, stream>>>(projo, meanb, stdb, outp);
}
